// BiRWKVLLADAAttention_10222022165144
// MI455X (gfx1250) — compile-verified, matrix operand order checked by reading, not run on hardware
//
#include <hip/hip_runtime.h>
#include <hip/hip_bf16.h>

// ---------------- problem constants (from reference) ----------------
#define B_ 4
#define T_ 2048
#define C_ 1024
#define TD_ 128

#define LCH 32            // chunk length along T
#define NCH (T_ / LCH)    // 64 chunks
#define BC_ (B_ * C_)     // 4096 channels

typedef __attribute__((ext_vector_type(16))) _Float16 v16h;
typedef __attribute__((ext_vector_type(8)))  float    v8f;

// =====================================================================
// Kernel 1: time-conditioning projection via WMMA (f16 in, f32 acc).
//   tmpW[b,c] = sum_k time_emb[b,k] * time_w_proj[c,k]   (ditto U)
//   decay[b,c] = exp(-exp(w[c] + tmpW*s)) * (0.5 + 0.5*tf[b])
//   ucond[b,c] = u[c] + tmpU*s
// One wave per 16-column tile of C (64 blocks x 32 threads).
// M dim = 16 (rows 0..3 = batch, rest zero via mask-multiply, NOT via
// divergent loads -- EXEC stays all-1s and loads stay coalesced).
// =====================================================================
__global__ void __launch_bounds__(32)
k_proj(const float* __restrict__ time_emb,   // [B, TD]
       const float* __restrict__ w,          // [C]
       const float* __restrict__ u,          // [C]
       const float* __restrict__ wproj,      // [C, TD]
       const float* __restrict__ uproj,      // [C, TD]
       const float* __restrict__ lts,        // [1]
       float* __restrict__ tf_out,           // [4]
       float* __restrict__ decay_out,        // [B*C]
       float* __restrict__ ucond_out)        // [B*C]
{
    const int lane = threadIdx.x;            // 0..31
    const int c0   = blockIdx.x * 16;        // column tile base
    const int m    = lane & 15;              // row (A) / col (B) index
    const int hi   = lane >> 4;              // lane half selects K sub-range
    const int mrow = m & 3;                  // clamped A row (always valid)
    const float amask = (m < 4) ? 1.f : 0.f; // zero-pad rows 4..15 by math

    // time factor tf[b] = sigmoid(sum_k time_emb[b,k]); cheap, per-lane.
    float tf4[4];
    #pragma unroll
    for (int b = 0; b < 4; ++b) {
        float s = 0.f;
        for (int j = 0; j < TD_; ++j) s += time_emb[b * TD_ + j];
        tf4[b] = 1.f / (1.f + __expf(-s));
    }
    if (blockIdx.x == 0 && lane < 4) tf_out[lane] = tf4[lane];

    v8f accW = {}; v8f accU = {};
    #pragma unroll
    for (int kc = 0; kc < TD_; kc += 32) {
        v16h a, bw, bu;
        // 16-bit A 16x32 layout (ISA 7.12.2): lanes 0-15 rows, VGPR v holds
        // K = {2v,2v+1} (v<4) or {16+2(v-4),...}; lanes 16-31 add +8 to K.
        #pragma unroll
        for (int e = 0; e < 16; ++e) {
            const int vg = e >> 1, h = e & 1;
            const int kk = (vg < 4) ? (hi * 8 + 2 * vg + h)
                                    : (16 + hi * 8 + 2 * (vg - 4) + h);
            a[e]  = (_Float16)(time_emb[mrow * TD_ + kc + kk] * amask);
            bw[e] = (_Float16)wproj[(c0 + m) * TD_ + kc + kk];
            bu[e] = (_Float16)uproj[(c0 + m) * TD_ + kc + kk];
        }
        accW = __builtin_amdgcn_wmma_f32_16x16x32_f16(
            false, a, false, bw, (short)0, accW, false, false);
        accU = __builtin_amdgcn_wmma_f32_16x16x32_f16(
            false, a, false, bu, (short)0, accU, false, false);
    }

    // D layout: VGPR r, lanes 0-15 -> M = r (0..7). Rows 0..3 are batches.
    const float scale = lts[0];
    if (hi == 0) {
        #pragma unroll
        for (int r = 0; r < 4; ++r) {
            const int b = r, c = c0 + m;
            const float wc  = w[c] + accW[r] * scale;
            const float dcy = __expf(-__expf(wc)) * (0.5f + 0.5f * tf4[b]);
            decay_out[b * C_ + c] = dcy;
            ucond_out[b * C_ + c] = u[c] + accU[r] * scale;
        }
    }
}

// =====================================================================
// Kernel 2: chunk-local bidirectional scan. One thread per (b, c, chunk).
// Backward local scan kept in a fully-unrolled register array, then the
// forward local scan fuses fwd + bwd + cross term into partial num/den.
// Chunk carries (full-chunk fwd sum G, bwd sum H) saved for fix-up.
// =====================================================================
__global__ void __launch_bounds__(256)
k_scan(const float* __restrict__ kin, const float* __restrict__ vin,
       const float* __restrict__ decay, const float* __restrict__ ucond,
       float* __restrict__ numP,           // [B*T*C]  (lives in d_out)
       float* __restrict__ denP,           // [B*T*C]
       float* __restrict__ Gn, float* __restrict__ Gd,
       float* __restrict__ Hn, float* __restrict__ Hd)  // [NCH][B*C]
{
    const int g  = blockIdx.x * blockDim.x + threadIdx.x;   // [0, B*C*NCH)
    const int c  = g & (C_ - 1);
    const int r  = g >> 10;            // C_ = 1024
    const int b  = r & (B_ - 1);
    const int ch = r >> 2;             // B_ = 4
    if (ch >= NCH) return;

    const int bc = b * C_ + c;
    const float d  = decay[bc];
    const float uc = ucond[bc];
    const int base = ((b * T_) + ch * LCH) * C_ + c;

    // ---- backward local scan (registers, full unroll) ----
    float bn[LCH], bd[LCH];
    float ab = 0.f, db = 0.f;
    #pragma unroll
    for (int tl = LCH - 1; tl >= 0; --tl) {
        const int idx = base + tl * C_;
        const float kk = kin[idx], vv = vin[idx];
        const float ek = __expf(fminf(kk, 30.f));
        ab = ab * d + ek * vv;
        db = db * d + ek;
        bn[tl] = ab; bd[tl] = db;
    }
    Hn[ch * BC_ + bc] = ab;
    Hd[ch * BC_ + bc] = db;

    // ---- forward local scan, fuse everything local ----
    float af = 0.f, df = 0.f;
    #pragma unroll
    for (int tl = 0; tl < LCH; ++tl) {
        const int idx = base + tl * C_;
        const float kk = kin[idx], vv = vin[idx];   // L0/L2 hit (re-read)
        const float ek  = __expf(fminf(kk, 30.f));
        const float euk = __expf(fminf(uc + kk, 30.f));
        af = af * d + ek * vv;
        df = df * d + ek;
        numP[idx] = af + bn[tl] + (euk - ek) * vv;
        denP[idx] = df + bd[tl] + (euk - ek);
    }
    Gn[ch * BC_ + bc] = af;
    Gd[ch * BC_ + bc] = df;
}

// =====================================================================
// Kernel 3: per-channel carry prefix (fwd) / suffix (bwd) over chunks.
//   Cf[i] = Cf[i-1]*d^L + G[i-1]   (Cf[0]=0)
//   Cb[i] = Cb[i+1]*d^L + H[i+1]   (Cb[NCH-1]=0)
// =====================================================================
__global__ void __launch_bounds__(256)
k_carry(const float* __restrict__ decay,
        const float* __restrict__ Gn, const float* __restrict__ Gd,
        const float* __restrict__ Hn, const float* __restrict__ Hd,
        float* __restrict__ CfN, float* __restrict__ CfD,
        float* __restrict__ CbN, float* __restrict__ CbD)
{
    const int g = blockIdx.x * blockDim.x + threadIdx.x;   // [0, B*C)
    if (g >= BC_) return;
    const float d  = decay[g];
    const float lg = __log2f(d);                 // d>0; underflow -> -inf ok
    const float dL = exp2f((float)LCH * lg);     // d^L

    float cn = 0.f, cd = 0.f;
    for (int i = 0; i < NCH; ++i) {
        CfN[i * BC_ + g] = cn; CfD[i * BC_ + g] = cd;
        cn = cn * dL + Gn[i * BC_ + g];
        cd = cd * dL + Gd[i * BC_ + g];
    }
    cn = 0.f; cd = 0.f;
    for (int i = NCH - 1; i >= 0; --i) {
        CbN[i * BC_ + g] = cn; CbD[i * BC_ + g] = cd;
        cn = cn * dL + Hn[i * BC_ + g];
        cd = cd * dL + Hd[i * BC_ + g];
    }
}

// =====================================================================
// Kernel 4: apply carries + final elementwise math.
//   num(t) = numP(t) + CfN*d^(tl+1) + CbN*d^(L-tl)    (den likewise)
//   out    = sigmoid(r) * num/max(den,1e-8) * (0.8+0.2*tf[b])
// =====================================================================
__global__ void __launch_bounds__(256)
k_final(const float* __restrict__ rin, const float* __restrict__ tf,
        const float* __restrict__ decay, const float* __restrict__ denP,
        const float* __restrict__ CfN, const float* __restrict__ CfD,
        const float* __restrict__ CbN, const float* __restrict__ CbD,
        float* __restrict__ out)             // in: numP, out: final
{
    const int g  = blockIdx.x * blockDim.x + threadIdx.x;   // [0, B*C*NCH)
    const int c  = g & (C_ - 1);
    const int r  = g >> 10;
    const int b  = r & (B_ - 1);
    const int ch = r >> 2;
    if (ch >= NCH) return;

    const int bc = b * C_ + c;
    const float lg  = __log2f(decay[bc]);
    const float tfb = tf[b];
    const float cfn = CfN[ch * BC_ + bc], cfd = CfD[ch * BC_ + bc];
    const float cbn = CbN[ch * BC_ + bc], cbd = CbD[ch * BC_ + bc];
    const int base = ((b * T_) + ch * LCH) * C_ + c;

    #pragma unroll 4
    for (int tl = 0; tl < LCH; ++tl) {
        const int idx = base + tl * C_;
        if (tl + 8 < LCH)  // stage upcoming lines (global_prefetch_b8)
            __builtin_prefetch(&rin[idx + 8 * C_], 0, 0);
        const float ff = exp2f((float)(tl + 1)   * lg);   // d^(tl+1)
        const float fb = exp2f((float)(LCH - tl) * lg);   // d^(L-tl)
        const float num = out[idx]  + cfn * ff + cbn * fb;
        const float den = denP[idx] + cfd * ff + cbd * fb;
        const float sr  = 1.f / (1.f + __expf(-rin[idx]));
        const float bse = sr * num / fmaxf(den, 1e-8f);
        out[idx] = bse * (0.8f + 0.2f * tfb);
    }
}

// =====================================================================
// Launcher
// =====================================================================
extern "C" void kernel_launch(void* const* d_in, const int* in_sizes, int n_in,
                              void* d_out, int out_size, void* d_ws, size_t ws_size,
                              hipStream_t stream) {
    const float* r_in      = (const float*)d_in[0];   // [B,T,C]
    const float* k_in      = (const float*)d_in[1];   // [B,T,C]
    const float* v_in      = (const float*)d_in[2];   // [B,T,C]
    const float* time_emb  = (const float*)d_in[3];   // [B,TD]
    const float* w_in      = (const float*)d_in[4];   // [1,1,C]
    const float* u_in      = (const float*)d_in[5];   // [1,1,C]
    const float* wproj     = (const float*)d_in[6];   // [C,TD]
    const float* uproj     = (const float*)d_in[7];   // [C,TD]
    const float* lts       = (const float*)d_in[8];   // [1]
    float* out = (float*)d_out;                       // [B,T,C] fp32

    // workspace carve-up (floats)
    float* ws   = (float*)d_ws;
    float* tf    = ws;                    // 8 (4 used)
    float* decay = tf + 8;                // B*C
    float* ucond = decay + BC_;           // B*C
    float* Gn    = ucond + BC_;           // NCH*B*C each below
    float* Gd    = Gn  + NCH * BC_;
    float* Hn    = Gd  + NCH * BC_;
    float* Hd    = Hn  + NCH * BC_;
    float* CfN   = Hd  + NCH * BC_;
    float* CfD   = CfN + NCH * BC_;
    float* CbN   = CfD + NCH * BC_;
    float* CbD   = CbN + NCH * BC_;
    float* denP  = CbD + NCH * BC_;       // B*T*C

    // 1) WMMA time projection -> tf, decay, ucond
    k_proj<<<C_ / 16, 32, 0, stream>>>(time_emb, w_in, u_in, wproj, uproj, lts,
                                       tf, decay, ucond);

    // 2) chunk-local bidirectional scans (B*C*NCH threads)
    const int nScan = BC_ * NCH;
    k_scan<<<nScan / 256, 256, 0, stream>>>(k_in, v_in, decay, ucond,
                                            out, denP, Gn, Gd, Hn, Hd);

    // 3) carry prefix/suffix per channel
    k_carry<<<BC_ / 256, 256, 0, stream>>>(decay, Gn, Gd, Hn, Hd,
                                           CfN, CfD, CbN, CbD);

    // 4) apply carries + finalize
    k_final<<<nScan / 256, 256, 0, stream>>>(r_in, tf, decay, denP,
                                             CfN, CfD, CbN, CbD, out);
}
